// Net_26594437497532
// MI455X (gfx1250) — matrix operand order repaired and checked by reading, not compiled, not run on hardware
//
#include <hip/hip_runtime.h>

// ===========================================================================
// MuseMorphose-style VAE transformer forward for MI455X (gfx1250, wave32).
// All dense math runs through v_wmma_f32_16x16x32_bf16 (fp32 -> bf16 convert
// on LDS staging, fp32 accumulate). GEMM: 128x64 block tile, 8 waves,
// double-buffered LDS with register-staged global loads so HBM latency hides
// under WMMA issue. fp32->bf16 packing uses v_perm_b32 (1 op per pair after
// round-bias add). Attention is flash-style (online softmax) with WMMA for
// both QK^T and P*V; the 1280x1280 decoder score matrix never materializes.
// ===========================================================================

typedef __attribute__((ext_vector_type(16))) __bf16 v16bf;
typedef __attribute__((ext_vector_type(8)))  float  v8f;
typedef __attribute__((ext_vector_type(4)))  float  fx4;   // native vec for b128 loads

// fp32 -> bf16 with round-half-up (bias 0x8000), single-value form
static __device__ __forceinline__ __bf16 f2bf(float f) {
  unsigned u = __builtin_bit_cast(unsigned, f) + 0x8000u;
  return __builtin_bit_cast(__bf16, (unsigned short)(u >> 16));
}

// pack two fp32 -> two bf16 in one dword via v_perm_b32 (lo->[15:0], hi->[31:16])
static __device__ __forceinline__ unsigned pack2bf(float lo, float hi) {
  unsigned a = __builtin_bit_cast(unsigned, lo) + 0x8000u;
  unsigned b = __builtin_bit_cast(unsigned, hi) + 0x8000u;
  // byte pool = {src0=b bytes 7..4, src1=a bytes 3..0}; pick b[3:2] | a[3:2]
  return __builtin_amdgcn_perm(b, a, 0x07060302u);
}

static __device__ __forceinline__ v8f wmma_bf16(v16bf a, v16bf b, v8f c) {
  // 8 args: (neg_a, A, neg_b, B, c_mod, C, reuse_a, reuse_b)
  return __builtin_amdgcn_wmma_f32_16x16x32_bf16(false, a, false, b,
                                                 (short)0, c, false, false);
}

// ISA 16-bit A-matrix 16x32 layout: lane (0-15 / 16-31) holds row m=lane&15;
// lane-half selects K sub-ranges {0-7,16-23} vs {8-15,24-31}.
static __device__ __forceinline__ int kmapA(int e, int hl) {
  return (e < 8) ? (hl * 8 + e) : (16 + hl * 8 + (e - 8));
}

// ---------------------------------------------------------------------------
// GEMM: C[M,N] = A[M,K] @ B[N,K]^T + bias (+ReLU). K multiple of 32.
// Block: 256 threads (8 waves), 128x64 tile; each wave a 32x32 sub-tile
// (waves laid out 4x2). Double-buffered LDS, register-staged global loads.
// ---------------------------------------------------------------------------
__global__ __launch_bounds__(256)
void gemm_bf16_wmma(const float* __restrict__ A, const float* __restrict__ B,
                    const float* __restrict__ bias, float* __restrict__ C,
                    int M, int N, int K, int relu)
{
  __shared__ __bf16 As[2][128][34];  // 17-bank row stride: conflict-free gathers
  __shared__ __bf16 Bs[2][64][34];
  const int tid  = threadIdx.x;
  const int lane = tid & 31;
  const int w    = tid >> 5;          // 0..7
  const int wm   = w >> 1;            // 0..3 (M direction)
  const int wn   = w & 1;             // 0..1 (N direction)
  const int hl   = lane >> 4, nn = lane & 15;
  const int bm   = blockIdx.y * 128, bn = blockIdx.x * 64;
  const int nk   = K >> 5;

  v8f acc[2][2] = {};
  fx4 fa[4], fb[2];                   // register staging for the next k-tile

  // ---- issue global loads for k-tile t (no wait; masking deferred) ----
  auto load_tile = [&](int t) {
    const int k0 = t << 5;
#pragma unroll
    for (int j = 0; j < 4; ++j) {     // A: 128x32 = 1024 fx4, 4 per thread
      int idx = j * 256 + tid;
      int r = idx >> 3, c4 = (idx & 7) << 2;
      int ar = bm + r;
      int arc = (ar < M) ? ar : (M - 1);
      fa[j] = *(const fx4*)(A + (size_t)arc * K + k0 + c4);
    }
#pragma unroll
    for (int j = 0; j < 2; ++j) {     // B: 64x32 = 512 fx4, 2 per thread
      int idx = j * 256 + tid;
      int r = idx >> 3, c4 = (idx & 7) << 2;
      int br = bn + r;
      int brc = (br < N) ? br : (N - 1);
      fb[j] = __builtin_nontemporal_load(
                  (const fx4*)(B + (size_t)brc * K + k0 + c4));
    }
  };

  // ---- mask + convert + store staged registers into LDS buffer s ----
  auto store_tile = [&](int s) {
#pragma unroll
    for (int j = 0; j < 4; ++j) {
      int idx = j * 256 + tid;
      int r = idx >> 3, c4 = (idx & 7) << 2;
      fx4 v = fa[j] * ((bm + r < M) ? 1.f : 0.f);
      *(unsigned*)&As[s][r][c4]     = pack2bf(v.x, v.y);
      *(unsigned*)&As[s][r][c4 + 2] = pack2bf(v.z, v.w);
    }
#pragma unroll
    for (int j = 0; j < 2; ++j) {
      int idx = j * 256 + tid;
      int r = idx >> 3, c4 = (idx & 7) << 2;
      fx4 v = fb[j] * ((bn + r < N) ? 1.f : 0.f);
      *(unsigned*)&Bs[s][r][c4]     = pack2bf(v.x, v.y);
      *(unsigned*)&Bs[s][r][c4 + 2] = pack2bf(v.z, v.w);
    }
  };

  load_tile(0);
  store_tile(0);

  for (int t = 0; t < nk; ++t) {
    __syncthreads();                      // buffer t&1 ready; (t+1)&1 free
    if (t + 1 < nk) load_tile(t + 1);     // loads in flight during WMMAs

    const int s = t & 1;
    v16bf av[2], bv[2];
#pragma unroll
    for (int e = 0; e < 16; ++e) {
      int ka = kmapA(e, hl);
      av[0][e] = As[s][wm * 32 +      nn][ka];
      av[1][e] = As[s][wm * 32 + 16 + nn][ka];
      int kb = kmapA(e, hl);              // B layout: n=nn, k=kmapA(e,hl)
      bv[0][e] = Bs[s][wn * 32 +      nn][kb];
      bv[1][e] = Bs[s][wn * 32 + 16 + nn][kb];
    }
#pragma unroll
    for (int mi = 0; mi < 2; ++mi)
#pragma unroll
      for (int ni = 0; ni < 2; ++ni)
        acc[mi][ni] = wmma_bf16(av[mi], bv[ni], acc[mi][ni]);

    if (t + 1 < nk) store_tile((t + 1) & 1);
  }

  // Hoisted bias (one predicated read per ni, not per element)
  float bvals[2] = {0.f, 0.f};
  if (bias) {
#pragma unroll
    for (int ni = 0; ni < 2; ++ni) {
      int n = bn + wn * 32 + ni * 16 + nn;
      if (n < N) bvals[ni] = bias[n];
    }
  }

#pragma unroll
  for (int mi = 0; mi < 2; ++mi)
#pragma unroll
    for (int ni = 0; ni < 2; ++ni) {
      int n = bn + wn * 32 + ni * 16 + nn;
      if (n < N) {
#pragma unroll
        for (int i = 0; i < 8; ++i) {       // C layout: m=i+8*hl, n=nn
          int m = bm + wm * 32 + mi * 16 + i + 8 * hl;
          if (m < M) {
            float v = acc[mi][ni][i] + bvals[ni];
            if (relu) v = fmaxf(v, 0.f);
            C[(size_t)m * N + n] = v;
          }
        }
      }
    }
}

// ---------------------------------------------------------------------------
// Flash attention, one wave per (16-query block, batch, head). hd = 64, H = 8.
// qkv rows: [token, 1536] = [Q(512) | K(512) | V(512)], token = s*B + b.
// out rows: [token, 512], head-concatenated.
// ---------------------------------------------------------------------------
__global__ __launch_bounds__(32)
void flash_attn_wmma(const float* __restrict__ qkv, float* __restrict__ out,
                     int S, int B, int H, int causal)
{
  __shared__ __bf16 qs[16][66];     // queries (33-bank stride)
  __shared__ __bf16 ks[32][66];     // key block
  __shared__ __bf16 vts[64][34];    // V^T block (d-major, 17-bank stride)
  __shared__ __bf16 ps[16][34];     // softmax probs

  const int lane = threadIdx.x;
  const int qb = blockIdx.x, bh = blockIdx.y;
  const int b = bh / H, h = bh % H;
  const int rs = 3 * 512;
  const int qcol = h * 64, kcol = 512 + h * 64, vcol = 1024 + h * 64;
  const int hl = lane >> 4, nn = lane & 15;

  // Stage Q 16x64: 256 float4, 8 per lane (b128 loads)
#pragma unroll
  for (int j = 0; j < 8; ++j) {
    int idx = j * 32 + lane;
    int m = idx >> 4, c4 = (idx & 15) << 2;
    fx4 f = *(const fx4*)(qkv + (size_t)((qb * 16 + m) * B + b) * rs
                              + qcol + c4);
    *(unsigned*)&qs[m][c4]     = pack2bf(f.x, f.y);
    *(unsigned*)&qs[m][c4 + 2] = pack2bf(f.z, f.w);
  }
  __syncthreads();

  v8f o0 = {}, o1 = {}, o2 = {}, o3 = {};
  float mrow[8], lrow[8];
#pragma unroll
  for (int i = 0; i < 8; ++i) { mrow[i] = -3.0e38f; lrow[i] = 0.f; }

  int nkb = S >> 5;
  if (causal) {
    int need = ((qb * 16 + 15) >> 5) + 1;
    if (need < nkb) nkb = need;
  }

  for (int kb = 0; kb < nkb; ++kb) {
    // Stage K (row-major) and V^T: 512 float4 each, 16 per lane
#pragma unroll 4
    for (int j = 0; j < 16; ++j) {
      int idx = j * 32 + lane;
      int kk = idx >> 4, c4 = (idx & 15) << 2;
      const float* row = qkv + (size_t)((kb * 32 + kk) * B + b) * rs;
      fx4 fk = *(const fx4*)(row + kcol + c4);
      fx4 fv = *(const fx4*)(row + vcol + c4);
      *(unsigned*)&ks[kk][c4]     = pack2bf(fk.x, fk.y);
      *(unsigned*)&ks[kk][c4 + 2] = pack2bf(fk.z, fk.w);
      vts[c4 + 0][kk] = f2bf(fv.x);
      vts[c4 + 1][kk] = f2bf(fv.y);
      vts[c4 + 2][kk] = f2bf(fv.z);
      vts[c4 + 3][kk] = f2bf(fv.w);
    }
    __syncthreads();

    // S_tile[16 x 32] = Q @ K^T : contract hd=64 in two K=32 WMMA steps
    v8f s0 = {}, s1 = {};
#pragma unroll
    for (int db = 0; db < 2; ++db) {
      v16bf a, bk0, bk1;
#pragma unroll
      for (int e = 0; e < 16; ++e) {
        a[e] = qs[nn][db * 32 + kmapA(e, hl)];
        int kd = db * 32 + kmapA(e, hl);
        bk0[e] = ks[     nn][kd];
        bk1[e] = ks[16 + nn][kd];
      }
      s0 = wmma_bf16(a, bk0, s0);
      s1 = wmma_bf16(a, bk1, s1);
    }

    // online softmax: row stats live across 16-lane groups (C layout rows)
#pragma unroll
    for (int i = 0; i < 8; ++i) {
      int m = qb * 16 + i + 8 * hl;
      float x0 = s0[i] * 0.125f;            // (hd=64)^-0.5
      float x1 = s1[i] * 0.125f;
      if (causal) {
        if (kb * 32 +      nn > m) x0 = -3.0e38f;
        if (kb * 32 + 16 + nn > m) x1 = -3.0e38f;
      }
      float rmax = fmaxf(x0, x1);
#pragma unroll
      for (int mk = 1; mk < 16; mk <<= 1) rmax = fmaxf(rmax, __shfl_xor(rmax, mk, 32));
      float nm = fmaxf(mrow[i], rmax);
      float p0 = __expf(x0 - nm);
      float p1 = __expf(x1 - nm);
      float rsum = p0 + p1;
#pragma unroll
      for (int mk = 1; mk < 16; mk <<= 1) rsum += __shfl_xor(rsum, mk, 32);
      float corr = __expf(mrow[i] - nm);
      lrow[i] = lrow[i] * corr + rsum;
      mrow[i] = nm;
      o0[i] *= corr; o1[i] *= corr; o2[i] *= corr; o3[i] *= corr;
      ps[i + 8 * hl][     nn] = f2bf(p0);
      ps[i + 8 * hl][16 + nn] = f2bf(p1);
    }
    __syncthreads();

    // O[16 x 64] += P[16 x 32] @ V[32 x 64]
    v16bf pa, bv0, bv1, bv2, bv3;
#pragma unroll
    for (int e = 0; e < 16; ++e) {
      pa[e] = ps[nn][kmapA(e, hl)];
      int kk = kmapA(e, hl);
      bv0[e] = vts[ 0 + nn][kk];
      bv1[e] = vts[16 + nn][kk];
      bv2[e] = vts[32 + nn][kk];
      bv3[e] = vts[48 + nn][kk];
    }
    o0 = wmma_bf16(pa, bv0, o0);
    o1 = wmma_bf16(pa, bv1, o1);
    o2 = wmma_bf16(pa, bv2, o2);
    o3 = wmma_bf16(pa, bv3, o3);
    __syncthreads();
  }

#pragma unroll
  for (int i = 0; i < 8; ++i) {
    float inv = 1.f / lrow[i];
    int m = qb * 16 + i + 8 * hl;
    float* orow = out + (size_t)(m * B + b) * 512 + h * 64;
    orow[ 0 + nn] = o0[i] * inv;
    orow[16 + nn] = o1[i] * inv;
    orow[32 + nn] = o2[i] * inv;
    orow[48 + nn] = o3[i] * inv;
  }
}

// ---------------------------------------------------------------------------
// LayerNorm(x + r) in-place over D=512. One 128-thread block per row.
// ---------------------------------------------------------------------------
__global__ __launch_bounds__(128)
void ln_residual(float* __restrict__ x, const float* __restrict__ r,
                 const float* __restrict__ g, const float* __restrict__ be)
{
  __shared__ float red[8];
  const int row = blockIdx.x, t = threadIdx.x;
  float v[4];
  float s = 0.f;
#pragma unroll
  for (int j = 0; j < 4; ++j) {
    int d = t + j * 128;
    v[j] = x[(size_t)row * 512 + d] + r[(size_t)row * 512 + d];
    s += v[j];
  }
#pragma unroll
  for (int o = 16; o > 0; o >>= 1) s += __shfl_down(s, o, 32);
  if ((t & 31) == 0) red[t >> 5] = s;
  __syncthreads();
  if (t == 0) red[4] = (red[0] + red[1] + red[2] + red[3]) * (1.f / 512.f);
  __syncthreads();
  const float mean = red[4];
  float q = 0.f;
#pragma unroll
  for (int j = 0; j < 4; ++j) { float dv = v[j] - mean; q += dv * dv; }
#pragma unroll
  for (int o = 16; o > 0; o >>= 1) q += __shfl_down(q, o, 32);
  if ((t & 31) == 0) red[t >> 5] = q;
  __syncthreads();
  if (t == 0) red[5] = rsqrtf((red[0] + red[1] + red[2] + red[3]) * (1.f / 512.f) + 1e-5f);
  __syncthreads();
  const float rstd = red[5];
#pragma unroll
  for (int j = 0; j < 4; ++j) {
    int d = t + j * 128;
    x[(size_t)row * 512 + d] = (v[j] - mean) * rstd * g[d] + be[d];
  }
}

// ---------------------------------------------------------------------------
// Small elementwise / gather kernels
// ---------------------------------------------------------------------------
static __device__ __forceinline__ float pe_val(int s, int d) {
  float i2  = (float)(d & ~1);  // 2*i
  float ang = (float)s * __expf(-(i2 / 512.f) * 9.210340371976184f); // ln(1e4)
  return (d & 1) ? __cosf(ang) : __sinf(ang);
}

__global__ void embed_enc_kernel(const float* __restrict__ tok_emb,
                                 const int* __restrict__ inp, float* __restrict__ X)
{
  int idx = blockIdx.x * blockDim.x + threadIdx.x;  // 8192*512
  int token = idx >> 9, d = idx & 511;
  int s = token >> 6, bb = token & 63;              // bb = b*16 + bar
  int tk = inp[s * 64 + bb];
  X[idx] = tok_emb[tk * 512 + d] * 22.62741699796952f + pe_val(s, d); // sqrt(512)
}

__global__ void embed_dec_kernel(const float* __restrict__ tok_emb,
                                 const int* __restrict__ inp, float* __restrict__ Y)
{
  int idx = blockIdx.x * blockDim.x + threadIdx.x;  // 5120*512
  int token = idx >> 9, d = idx & 511;
  int s = token >> 2, b = token & 3;
  int tk = inp[s * 4 + b];
  Y[idx] = tok_emb[tk * 512 + d] * 22.62741699796952f + pe_val(s, d);
}

__global__ void z_kernel(const float* __restrict__ eps, const float* __restrict__ mu,
                         const float* __restrict__ lv, float* __restrict__ z, int n)
{
  int i = blockIdx.x * blockDim.x + threadIdx.x;
  if (i < n) z[i] = eps[i] * __expf(0.5f * lv[i]) + mu[i];
}

__global__ void add_kernel(const float* __restrict__ a, const float* __restrict__ b,
                           float* __restrict__ c, int n)
{
  int i = blockIdx.x * blockDim.x + threadIdx.x;
  if (i < n) c[i] = a[i] + b[i];
}

// seg_cat[token, 0:128]=z_r[b,bar], [128:160)=rf, [160:192)=poly, [192:224)=style
__global__ void segcat_kernel(const float* __restrict__ z, const int* __restrict__ bar_pos,
                              const float* __restrict__ rfE, const float* __restrict__ plE,
                              const float* __restrict__ stE, const int* __restrict__ rf,
                              const int* __restrict__ pl, const int* __restrict__ st,
                              float* __restrict__ SC)
{
  int idx = blockIdx.x * blockDim.x + threadIdx.x;
  if (idx >= 5120 * 224) return;
  int token = idx / 224, c = idx % 224;
  int s = token >> 2, b = token & 3;
  float v;
  if (c < 128) {
    const int* pos = bar_pos + b * 17;
    int bid = 0;
#pragma unroll
    for (int j = 1; j < 17; ++j) bid += (pos[j] <= s) ? 1 : 0;  // searchsorted-right - 1
    v = z[(b * 16 + bid) * 128 + c];
  } else if (c < 160) {
    v = rfE[rf[s * 4 + b] * 32 + (c - 128)] * 5.656854249492381f;   // sqrt(32)
  } else if (c < 192) {
    v = plE[pl[s * 4 + b] * 32 + (c - 160)] * 5.656854249492381f;
  } else {
    v = stE[st[s * 4 + b] * 32 + (c - 192)] * 5.656854249492381f;
  }
  SC[idx] = v;
}

// ---------------------------------------------------------------------------
// Host orchestration
// ---------------------------------------------------------------------------
struct LayerStack {
  const float *Wqkv, *bqkv, *Wo, *bo, *W1, *b1, *W2, *b2, *g1, *be1, *g2, *be2;
};

static inline void launch_gemm(const float* A, const float* B, const float* bias,
                               float* C, int M, int N, int K, int relu,
                               hipStream_t st)
{
  dim3 grid((N + 63) / 64, (M + 127) / 128);
  gemm_bf16_wmma<<<grid, 256, 0, st>>>(A, B, bias, C, M, N, K, relu);
}

extern "C" void kernel_launch(void* const* d_in, const int* in_sizes, int n_in,
                              void* d_out, int out_size, void* d_ws, size_t ws_size,
                              hipStream_t stream)
{
  (void)in_sizes; (void)n_in; (void)out_size; (void)ws_size;

  const float* tok_emb = (const float*)d_in[0];
  LayerStack enc = {
    (const float*)d_in[1],  (const float*)d_in[2],  (const float*)d_in[3],
    (const float*)d_in[4],  (const float*)d_in[5],  (const float*)d_in[6],
    (const float*)d_in[7],  (const float*)d_in[8],  (const float*)d_in[9],
    (const float*)d_in[10], (const float*)d_in[11], (const float*)d_in[12] };
  const float* fc_mu_W = (const float*)d_in[13];
  const float* fc_mu_b = (const float*)d_in[14];
  const float* fc_lv_W = (const float*)d_in[15];
  const float* fc_lv_b = (const float*)d_in[16];
  LayerStack dec = {
    (const float*)d_in[17], (const float*)d_in[18], (const float*)d_in[19],
    (const float*)d_in[20], (const float*)d_in[21], (const float*)d_in[22],
    (const float*)d_in[23], (const float*)d_in[24], (const float*)d_in[25],
    (const float*)d_in[26], (const float*)d_in[27], (const float*)d_in[28] };
  const float* seg_W   = (const float*)d_in[29];
  const float* rfE     = (const float*)d_in[30];
  const float* plE     = (const float*)d_in[31];
  const float* stE     = (const float*)d_in[32];
  const float* out_W   = (const float*)d_in[33];
  const float* out_b   = (const float*)d_in[34];
  const float* eps     = (const float*)d_in[35];
  const int*   enc_inp = (const int*)d_in[36];
  const int*   dec_inp = (const int*)d_in[37];
  const int*   bar_pos = (const int*)d_in[38];
  const int*   rf      = (const int*)d_in[39];
  const int*   pl      = (const int*)d_in[40];
  const int*   st      = (const int*)d_in[41];

  float* out    = (float*)d_out;
  float* mu_out = out;                 // [64,128]
  float* lv_out = out + 64 * 128;      // [64,128]
  float* logits = out + 2 * 64 * 128;  // [5120,336]

  // Workspace carve-up (~187 MB of f32)
  const int Me = 8192, Md = 5120;
  float* ws   = (float*)d_ws;
  float* X    = ws;  ws += (size_t)Me * 512;    // encoder stream
  float* QKV  = ws;  ws += (size_t)Me * 1536;   // shared enc/dec qkv
  float* Aout = ws;  ws += (size_t)Me * 512;    // attention out / ffn out
  float* Hbuf = ws;  ws += (size_t)Me * 2048;   // ffn hidden (Tproj aliases head)
  float* Y    = ws;  ws += (size_t)Md * 512;    // decoder stream
  float* Yin  = ws;  ws += (size_t)Md * 512;    // decoder stream (+seg)
  float* Seg  = ws;  ws += (size_t)Md * 512;    // seg projection
  float* SC   = ws;  ws += (size_t)Md * 224;    // seg concat
  float* Z    = ws;  ws += (size_t)64 * 128;    // latent z
  float* Tproj = Hbuf;                          // Wo output; freed before FF1 use

  // ---- Encoder ----
  embed_enc_kernel<<<(Me * 512) / 256, 256, 0, stream>>>(tok_emb, enc_inp, X);
  for (int i = 0; i < 12; ++i) {
    launch_gemm(X, enc.Wqkv + (size_t)i * 1536 * 512, enc.bqkv + i * 1536,
                QKV, Me, 1536, 512, 0, stream);
    flash_attn_wmma<<<dim3(128 / 16, 64 * 8), 32, 0, stream>>>(QKV, Aout, 128, 64, 8, 0);
    launch_gemm(Aout, enc.Wo + (size_t)i * 512 * 512, enc.bo + i * 512,
                Tproj, Me, 512, 512, 0, stream);
    ln_residual<<<Me, 128, 0, stream>>>(X, Tproj, enc.g1 + i * 512, enc.be1 + i * 512);
    launch_gemm(X, enc.W1 + (size_t)i * 2048 * 512, enc.b1 + i * 2048,
                Hbuf, Me, 2048, 512, 1, stream);
    launch_gemm(Hbuf, enc.W2 + (size_t)i * 512 * 2048, enc.b2 + i * 512,
                Aout, Me, 512, 2048, 0, stream);
    ln_residual<<<Me, 128, 0, stream>>>(X, Aout, enc.g2 + i * 512, enc.be2 + i * 512);
  }

  // ---- Pool (s=0 rows 0..63) -> mu / logvar -> z ----
  launch_gemm(X, fc_mu_W, fc_mu_b, mu_out, 64, 128, 512, 0, stream);
  launch_gemm(X, fc_lv_W, fc_lv_b, lv_out, 64, 128, 512, 0, stream);
  z_kernel<<<(8192 + 255) / 256, 256, 0, stream>>>(eps, mu_out, lv_out, Z, 8192);

  // ---- Decoder prep ----
  embed_dec_kernel<<<(Md * 512) / 256, 256, 0, stream>>>(tok_emb, dec_inp, Y);
  segcat_kernel<<<(Md * 224 + 255) / 256, 256, 0, stream>>>(Z, bar_pos, rfE, plE, stE,
                                                            rf, pl, st, SC);
  launch_gemm(SC, seg_W, nullptr, Seg, Md, 512, 224, 0, stream);

  // ---- In-attention decoder (causal) ----
  float* cur = Y;
  float* nxt = Yin;
  for (int i = 0; i < 12; ++i) {
    add_kernel<<<(Md * 512) / 256, 256, 0, stream>>>(cur, Seg, nxt, Md * 512);
    launch_gemm(nxt, dec.Wqkv + (size_t)i * 1536 * 512, dec.bqkv + i * 1536,
                QKV, Md, 1536, 512, 0, stream);
    flash_attn_wmma<<<dim3(1280 / 16, 4 * 8), 32, 0, stream>>>(QKV, Aout, 1280, 4, 8, 1);
    launch_gemm(Aout, dec.Wo + (size_t)i * 512 * 512, dec.bo + i * 512,
                Tproj, Md, 512, 512, 0, stream);
    ln_residual<<<Md, 128, 0, stream>>>(nxt, Tproj, dec.g1 + i * 512, dec.be1 + i * 512);
    launch_gemm(nxt, dec.W1 + (size_t)i * 2048 * 512, dec.b1 + i * 2048,
                Hbuf, Md, 2048, 512, 1, stream);
    launch_gemm(Hbuf, dec.W2 + (size_t)i * 512 * 2048, dec.b2 + i * 512,
                Aout, Md, 512, 2048, 0, stream);
    ln_residual<<<Md, 128, 0, stream>>>(nxt, Aout, dec.g2 + i * 512, dec.be2 + i * 512);
    float* t = cur; cur = nxt; nxt = t;       // layer output becomes new stream
  }

  // ---- Output projection [5120,512] x [336,512]^T ----
  launch_gemm(cur, out_W, out_b, logits, Md, 336, 512, 0, stream);
}
